// Block_1w1a_73280732004541
// MI455X (gfx1250) — 8-bit matrix operand order repaired and checked by reading, compiled, not run on hardware
//
#include <hip/hip_runtime.h>
#include <stdint.h>

typedef int  v8i  __attribute__((ext_vector_type(8)));
typedef int  v4i  __attribute__((ext_vector_type(4)));

#define BN   32
#define CCH  512
#define HH   56
#define WW   56
#define COO  512
#define HWSZ (HH * WW)
#define EPSV 1e-5f

#define LDS_STRIDE 528

__global__ __launch_bounds__(256) void bnn_prep_kernel(
    const float* __restrict__ w_dw, const float* __restrict__ w_pw,
    const float* __restrict__ g1, const float* __restrict__ b1,
    const float* __restrict__ m1, const float* __restrict__ v1,
    const float* __restrict__ g2, const float* __restrict__ b2,
    const float* __restrict__ m2, const float* __restrict__ v2,
    int8_t* __restrict__ Bw, float* __restrict__ inv1, float* __restrict__ shift1,
    float* __restrict__ inv2, float* __restrict__ shift2, uint32_t* __restrict__ dwbits)
{
    const int i = blockIdx.x * blockDim.x + threadIdx.x;
    if (i < COO * CCH)
        Bw[i] = (w_pw[i] >= 0.0f) ? (int8_t)1 : (int8_t)-1;
    if (i < CCH) {
        const float iv = g1[i] * rsqrtf(v1[i] + EPSV);
        inv1[i]   = iv;
        shift1[i] = b1[i] - m1[i] * iv;
        uint32_t bits = 0;
        for (int t = 0; t < 9; ++t)
            bits |= (w_dw[i * 9 + t] < 0.0f ? 1u : 0u) << t;
        dwbits[i] = bits;
    }
    if (i < COO) {
        const float iv = g2[i] * rsqrtf(v2[i] + EPSV);
        inv2[i]   = iv;
        shift2[i] = b2[i] - m2[i] * iv;
    }
}

__global__ __launch_bounds__(256) void bnn_block_kernel(
    const float*    __restrict__ x,
    const int8_t*   __restrict__ Bw,
    const float*    __restrict__ inv1, const float* __restrict__ shift1,
    const float*    __restrict__ inv2, const float* __restrict__ shift2,
    const uint32_t* __restrict__ dwbits,
    float*          __restrict__ out)
{
    __shared__ int8_t As[16 * LDS_STRIDE];

    const int tid = threadIdx.x;
    const int m0  = blockIdx.x * 16;
    const int b   = m0 / HWSZ;
    const int s0  = m0 - b * HWSZ;

    {
        const int sl = tid & 15;
        const int cl = tid >> 4;
        const int s  = s0 + sl;
        const int h  = s / WW;
        const int w  = s - h * WW;
        const float* xb = x + (size_t)b * CCH * HWSZ;

        for (int j = 0; j < 8; ++j) {
            const int cbase = cl * 4 + j * 64;
            uint32_t packed = 0;
            #pragma unroll
            for (int cc = 0; cc < 4; ++cc) {
                const int c = cbase + cc;
                const float* xc = xb + (size_t)c * HWSZ;
                const uint32_t wb = dwbits[c];
                int sum = 0;
                #pragma unroll
                for (int dy = -1; dy <= 1; ++dy) {
                    const int hh = h + dy;
                    if (hh < 0 || hh >= HH) continue;
                    #pragma unroll
                    for (int dx = -1; dx <= 1; ++dx) {
                        const int wx = w + dx;
                        if (wx < 0 || wx >= WW) continue;
                        const float xv = xc[hh * WW + wx];
                        const uint32_t sx = __float_as_uint(xv) >> 31;
                        const uint32_t sw = (wb >> ((dy + 1) * 3 + (dx + 1))) & 1u;
                        sum += 1 - 2 * (int)(sx ^ sw);
                    }
                }
                const float val = (float)sum * inv1[c] + shift1[c];
                const uint32_t a8 = (val >= 0.0f) ? 0x01u : 0xFFu;
                packed |= a8 << (cc * 8);
            }
            *(uint32_t*)&As[sl * LDS_STRIDE + cbase] = packed;
        }
    }
    __syncthreads();

    const int wave = tid >> 5;
    const int lane = tid & 31;
    const int lmod = lane & 15;
    const int lhi  = lane >> 4;

    v8i acc[4];
    #pragma unroll
    for (int t = 0; t < 4; ++t) { v8i z = {0,0,0,0,0,0,0,0}; acc[t] = z; }

    const int n0base = wave * 64;
    const int arow   = lmod * LDS_STRIDE;

    for (int kc = 0; kc < 8; ++kc) {
        const int k0 = kc * 64;

        v8i afrag;
        {
            const int abase = arow + k0 + lhi * 16;
            uint64_t q;
            q = *(const uint64_t*)&As[abase +  0]; afrag[0] = (int)q; afrag[1] = (int)(q >> 32);
            q = *(const uint64_t*)&As[abase +  8]; afrag[2] = (int)q; afrag[3] = (int)(q >> 32);
            q = *(const uint64_t*)&As[abase + 32]; afrag[4] = (int)q; afrag[5] = (int)(q >> 32);
            q = *(const uint64_t*)&As[abase + 40]; afrag[6] = (int)q; afrag[7] = (int)(q >> 32);
        }

        #pragma unroll
        for (int t = 0; t < 4; ++t) {
            const int n = n0base + t * 16 + lmod;
            const int8_t* brow = Bw + (size_t)n * CCH + k0 + lhi * 16;
            const v4i bl0 = *(const v4i*)(brow);
            const v4i bl1 = *(const v4i*)(brow + 32);
            v8i bfrag;
            bfrag[0] = bl0[0]; bfrag[1] = bl0[1]; bfrag[2] = bl0[2]; bfrag[3] = bl0[3];
            bfrag[4] = bl1[0]; bfrag[5] = bl1[1]; bfrag[6] = bl1[2]; bfrag[7] = bl1[3];

            acc[t] = __builtin_amdgcn_wmma_i32_16x16x64_iu8(
                 true, afrag,  true, bfrag, acc[t],
                 false,  false);
        }
    }

    const size_t outb = (size_t)b * COO * HWSZ;
    #pragma unroll
    for (int t = 0; t < 4; ++t) {
        const int n = n0base + t * 16 + lmod;
        const float iv = inv2[n], sh = shift2[n];
        float* op = out + outb + (size_t)n * HWSZ + (s0 + lhi * 8);
        float4 o0, o1;
        o0.x = (float)acc[t][0] * iv + sh;  o0.y = (float)acc[t][1] * iv + sh;
        o0.z = (float)acc[t][2] * iv + sh;  o0.w = (float)acc[t][3] * iv + sh;
        o1.x = (float)acc[t][4] * iv + sh;  o1.y = (float)acc[t][5] * iv + sh;
        o1.z = (float)acc[t][6] * iv + sh;  o1.w = (float)acc[t][7] * iv + sh;
        *(float4*)(op)     = o0;
        *(float4*)(op + 4) = o1;
    }
}

extern "C" void kernel_launch(void* const* d_in, const int* in_sizes, int n_in,
                              void* d_out, int out_size, void* d_ws, size_t ws_size,
                              hipStream_t stream) {
    (void)in_sizes; (void)n_in; (void)out_size; (void)ws_size;

    const float* x    = (const float*)d_in[0];
    const float* w_dw = (const float*)d_in[1];
    const float* w_pw = (const float*)d_in[2];
    const float* g1   = (const float*)d_in[3];
    const float* b1   = (const float*)d_in[4];
    const float* m1   = (const float*)d_in[5];
    const float* v1   = (const float*)d_in[6];
    const float* g2   = (const float*)d_in[7];
    const float* b2   = (const float*)d_in[8];
    const float* m2   = (const float*)d_in[9];
    const float* v2   = (const float*)d_in[10];
    float* out = (float*)d_out;

    char* ws = (char*)d_ws;
    int8_t*   Bw     = (int8_t*)ws;
    float*    inv1   = (float*)(ws + (size_t)COO * CCH);
    float*    shift1 = inv1   + CCH;
    float*    inv2   = shift1 + CCH;
    float*    shift2 = inv2   + COO;
    uint32_t* dwbits = (uint32_t*)(shift2 + COO);

    const int prep_n = COO * CCH;
    bnn_prep_kernel<<<(prep_n + 255) / 256, 256, 0, stream>>>(
        w_dw, w_pw, g1, b1, m1, v1, g2, b2, m2, v2,
        Bw, inv1, shift1, inv2, shift2, dwbits);

    const int ntiles = (BN * HWSZ) / 16;
    bnn_block_kernel<<<ntiles, 256, 0, stream>>>(
        x, Bw, inv1, shift1, inv2, shift2, dwbits, out);
}
